// MultiHeadAttention_15135464751648
// MI455X (gfx1250) — matrix operand order repaired and checked by reading, not compiled, not run on hardware
//
#include <hip/hip_runtime.h>
#include <hip/hip_bf16.h>

// ---------------------------------------------------------------------------
// MultiHeadAttention (RoPE + causal flash attention) for MI455X / gfx1250
//   x[2,2048,1024] f32, Wq/Wk/Wv/Wo[1024,1024] f32  ->  out[2,2048,1024] f32
// Matmuls via v_wmma_f32_16x16x32_f16; weight panels staged to LDS with the
// Tensor Data Mover (tensor_load_to_lds + s_wait_tensorcnt).
// ---------------------------------------------------------------------------

#define S_LEN   2048
#define DM      1024
#define NH      16
#define HD      64
#define MTOT    4096            // B * S

typedef _Float16 f16;
typedef _Float16 f16x8  __attribute__((ext_vector_type(8)));
typedef _Float16 f16x16 __attribute__((ext_vector_type(16)));
typedef float    v8f    __attribute__((ext_vector_type(8)));
typedef unsigned int u32x4 __attribute__((ext_vector_type(4)));
typedef int          i32x4 __attribute__((ext_vector_type(4)));
typedef int          i32x8 __attribute__((ext_vector_type(8)));

static __device__ __forceinline__ f16x16 cat8(f16x8 lo, f16x8 hi) {
  return __builtin_shufflevector(lo, hi, 0,1,2,3,4,5,6,7,8,9,10,11,12,13,14,15);
}

static __device__ __forceinline__ v8f wmma16(f16x16 a, f16x16 b, v8f c) {
  // D = A(16x32) * B(32x16) + C(16x16), f32 accumulate
  return __builtin_amdgcn_wmma_f32_16x16x32_f16(false, a, false, b,
                                                (short)0, c, false, false);
}

// ---- Tensor Data Mover: 2-D tile (f16) global -> LDS ----------------------
// D# per CDNA5 ISA ch.8: group0 = {count/type, lds_addr, global_addr},
// group1 = {data_size, tensor dims, tile dims, dim0 stride}. Higher groups
// zero (<=2D tensor). All variable dwords made wave-uniform via readfirstlane.
// This toolchain's builtin is the 6-arg form: (g0, g1, g2, g3, g_extra, cpol).
static __device__ __forceinline__ void tdm_load_2d_f16(
    unsigned lds_byte_off, const f16* gptr,
    unsigned tensor_w, unsigned tensor_h, unsigned stride_elems,
    unsigned tile_w, unsigned tile_h) {
  unsigned long long ga = (unsigned long long)(uintptr_t)gptr;
  unsigned ga_lo = __builtin_amdgcn_readfirstlane((unsigned)(ga & 0xFFFFFFFFu));
  unsigned ga_hi = __builtin_amdgcn_readfirstlane((unsigned)(ga >> 32));
  unsigned lds   = __builtin_amdgcn_readfirstlane(lds_byte_off);

  u32x4 g0;
  g0[0] = 1u;                                      // count=1, user descriptor
  g0[1] = lds;                                     // lds_addr (bytes)
  g0[2] = ga_lo;                                   // global_addr[31:0]
  g0[3] = (ga_hi & 0x1FFFFFFu) | (2u << 30);       // global_addr[56:32] | type=2

  i32x8 g1;
  g1[0] = (int)(1u << 16);                         // data_size=1 -> 2 bytes
  g1[1] = (int)((tensor_w & 0xFFFFu) << 16);       // tensor_dim0[15:0]
  g1[2] = (int)(((tensor_w >> 16) & 0xFFFFu) |     // tensor_dim0[31:16]
                ((tensor_h & 0xFFFFu) << 16));     // tensor_dim1[15:0]
  g1[3] = (int)(((tensor_h >> 16) & 0xFFFFu) |     // tensor_dim1[31:16]
                ((tile_w & 0xFFFFu) << 16));       // tile_dim0
  g1[4] = (int)(tile_h & 0xFFFFu);                 // tile_dim1 (tile_dim2=0)
  g1[5] = (int)stride_elems;                       // tensor_dim0_stride[31:0]
  g1[6] = 0;
  g1[7] = 0;
  i32x4 gz4 = {0, 0, 0, 0};
  i32x8 gz8 = {0, 0, 0, 0, 0, 0, 0, 0};
  __builtin_amdgcn_tensor_load_to_lds(g0, g1, gz4, gz4, gz8, 0);
}

// --------------------------- f32 -> f16 converts ---------------------------

__global__ __launch_bounds__(256) void cvt_f16_kernel(
    const float* __restrict__ src, f16* __restrict__ dst, int n) {
  int i = blockIdx.x * 256 + threadIdx.x;
  if (i < n) dst[i] = (f16)src[i];
}

// W[n][k] (1024x1024) -> Wt[k][n]  (K-major for WMMA B-fragments)
__global__ __launch_bounds__(256) void cvt_wt_kernel(
    const float* __restrict__ W, f16* __restrict__ Wt) {
  int idx = blockIdx.x * 256 + threadIdx.x;   // 1M threads
  int k = idx >> 10;
  int n = idx & 1023;
  Wt[idx] = (f16)W[n * DM + k];
}

// ------------------------------ GEMM mainloop ------------------------------
// Y[m,n] = sum_k X[m,k] * Wt[k,n]; workgroup = 256x64 tile, wave = 32x64.
// B panel TDM-staged to LDS in 2 phases of 512 K-rows (64 KB); A fragments
// double-buffered in registers to overlap global loads with WMMA.
#define KPH 512                      // K rows per LDS phase

static __device__ __forceinline__ void gemm_panel(
    const f16* __restrict__ X, const f16* __restrict__ Wt,
    f16* __restrict__ Bs,            // LDS: [KPH][64] f16
    int m0, int n0, v8f acc[2][4]) {
  const int wave = threadIdx.x >> 5;
  const int lane = threadIdx.x & 31;
  const int ll = lane & 15;
  const int lh = lane >> 4;
  const int kb8 = lh * 8;
  const unsigned lds_base = (unsigned)(uintptr_t)Bs;

  for (int ph = 0; ph < 2; ++ph) {
    // Each wave DMAs 64 K-rows of the 64-wide N panel into its LDS slice.
    tdm_load_2d_f16(lds_base + (unsigned)wave * 64 * 128,
                    Wt + ((size_t)(ph * KPH + wave * 64)) * DM + n0,
                    DM, DM, DM, 64, 64);
    __builtin_amdgcn_s_wait_tensorcnt(0);
    __syncthreads();

    // Preload A fragments for k=0 of this phase.
    f16x16 af[2], afn[2];
#pragma unroll
    for (int i = 0; i < 2; ++i) {
      const f16* xp = X + (size_t)(m0 + 16 * i + ll) * DM + ph * KPH + kb8;
      af[i] = cat8(*(const f16x8*)xp, *(const f16x8*)(xp + 16));
    }
    for (int k = 0; k < KPH; k += 32) {
      if (k + 32 < KPH) {
#pragma unroll
        for (int i = 0; i < 2; ++i) {
          const f16* xp =
              X + (size_t)(m0 + 16 * i + ll) * DM + ph * KPH + k + 32 + kb8;
          afn[i] = cat8(*(const f16x8*)xp, *(const f16x8*)(xp + 16));
        }
      }
      const f16* bp = Bs + (size_t)(k + kb8) * 64 + ll;
#pragma unroll
      for (int j = 0; j < 4; ++j) {
        f16x16 bfr;
#pragma unroll
        for (int s = 0; s < 16; ++s)
          bfr[s] = bp[(size_t)((s & 7) + ((s >> 3) << 4)) * 64 + 16 * j];
#pragma unroll
        for (int i = 0; i < 2; ++i)
          acc[i][j] = wmma16(af[i], bfr, acc[i][j]);
      }
      if (k + 32 < KPH) {
        af[0] = afn[0];
        af[1] = afn[1];
      }
    }
    __syncthreads();   // phase data fully consumed before next TDM overwrite
  }
}

// ------------------------ QKV projection (+RoPE) ---------------------------
// blockIdx.z: 0 = Q (rope, scaled, [B,H,S,HD]), 1 = K (rope, [B,H,HD,S]),
//             2 = V (plain, [B,H,S,HD])
__global__ __launch_bounds__(256) void qkv_kernel(
    const f16* __restrict__ Xb,
    const f16* __restrict__ Wtq, const f16* __restrict__ Wtk,
    const f16* __restrict__ Wtv,
    f16* __restrict__ Qr, f16* __restrict__ Kt, f16* __restrict__ Vn) {
  __shared__ __attribute__((aligned(128))) f16 Bs[KPH * 64];   // 64 KB

  const int mode = blockIdx.z;
  const f16* Wt = (mode == 0) ? Wtq : ((mode == 1) ? Wtk : Wtv);
  const int wave = threadIdx.x >> 5;
  const int lane = threadIdx.x & 31;
  const int ll = lane & 15;
  const int lh = lane >> 4;
  const int m0 = blockIdx.x * 256 + wave * 32;
  const int n0 = blockIdx.y * 64;           // head-aligned (HD == 64)
  const int h  = n0 >> 6;

  v8f acc[2][4];
#pragma unroll
  for (int i = 0; i < 2; ++i)
#pragma unroll
    for (int j = 0; j < 4; ++j) acc[i][j] = (v8f)0.0f;

  gemm_panel(Xb, Wt, Bs, m0, n0, acc);

  if (mode < 2) {
    const float qscale = (mode == 0) ? 0.125f : 1.0f;  // 1/sqrt(HD) into Q
#pragma unroll
    for (int i = 0; i < 2; ++i) {
#pragma unroll
      for (int j = 0; j < 2; ++j) {
        const int dlo = 16 * j + ll;                     // 0..31
        const float inv = __expf(-9.2103403719761836f * (float)(2 * dlo) *
                                 (1.0f / 64.0f));        // 10000^(-2*d/64)
#pragma unroll
        for (int v = 0; v < 8; ++v) {
          const int m = m0 + 16 * i + v + 8 * lh;
          const int b = m >> 11;
          const int s = m & (S_LEN - 1);
          const float th = (float)s * inv;
          const float c = cosf(th), sn = sinf(th);
          const float a  = acc[i][j][v];
          const float bb = acc[i][j + 2][v];
          const float lo = (a * c - bb * sn) * qscale;
          const float hi = (bb * c + a * sn) * qscale;
          const size_t bh = (size_t)(b * NH + h);
          if (mode == 0) {
            f16* qp = Qr + (bh * S_LEN + s) * HD;
            qp[dlo] = (f16)lo;
            qp[dlo + 32] = (f16)hi;
          } else {
            f16* kp = Kt + bh * (size_t)HD * S_LEN + s;
            kp[(size_t)dlo * S_LEN] = (f16)lo;
            kp[(size_t)(dlo + 32) * S_LEN] = (f16)hi;
          }
        }
      }
    }
  } else {
#pragma unroll
    for (int i = 0; i < 2; ++i)
#pragma unroll
      for (int j = 0; j < 4; ++j)
#pragma unroll
        for (int v = 0; v < 8; ++v) {
          const int m = m0 + 16 * i + v + 8 * lh;
          const int b = m >> 11;
          const int s = m & (S_LEN - 1);
          const int d = 16 * j + ll;
          Vn[((size_t)(b * NH + h) * S_LEN + s) * HD + d] = (f16)acc[i][j][v];
        }
  }
}

// ----------------------- Flash attention (causal) --------------------------
// grid = (S/128, B*NH), 256 threads; wave w owns 16 query rows.
__global__ __launch_bounds__(256) void attn_kernel(
    const f16* __restrict__ Qr, const f16* __restrict__ Ktr,
    const f16* __restrict__ Vn, f16* __restrict__ Ob) {
  __shared__ f16 Pl[8][16 * 64];   // per-wave P staging (C-layout -> A-layout)

  const int wave = threadIdx.x >> 5;
  const int lane = threadIdx.x & 31;
  const int ll = lane & 15;
  const int lh = lane >> 4;
  const int kb8 = lh * 8;
  const int bh = blockIdx.y;
  const int b = bh >> 4;
  const int h = bh & (NH - 1);
  const int q0 = blockIdx.x * 128 + wave * 16;

  const f16* Q  = Qr  + (size_t)bh * S_LEN * HD;
  const f16* Kt = Ktr + (size_t)bh * HD * S_LEN;
  const f16* V  = Vn  + (size_t)bh * S_LEN * HD;
  f16* Pw = &Pl[wave][0];

  // Q A-fragments for this wave's 16 rows (held for the whole key loop)
  f16x16 qf[2];
#pragma unroll
  for (int t = 0; t < 2; ++t) {
    const f16* qp = Q + (size_t)(q0 + ll) * HD + 32 * t + kb8;
    qf[t] = cat8(*(const f16x8*)qp, *(const f16x8*)(qp + 16));
  }

  float row_m[8], row_l[8];
  v8f oacc[4];
#pragma unroll
  for (int v = 0; v < 8; ++v) { row_m[v] = -1e30f; row_l[v] = 0.0f; }
#pragma unroll
  for (int jd = 0; jd < 4; ++jd) oacc[jd] = (v8f)0.0f;

  for (int kb = 0; kb <= q0 + 15; kb += 64) {
    // ---- scores: S = Q * K^T over a 16x64 key block ----
    v8f sc[4];
#pragma unroll
    for (int j = 0; j < 4; ++j) sc[j] = (v8f)0.0f;
#pragma unroll
    for (int t = 0; t < 2; ++t) {
      const f16* kp = Kt + (size_t)(32 * t + kb8) * S_LEN + kb + ll;
#pragma unroll
      for (int j = 0; j < 4; ++j) {
        f16x16 bK;
#pragma unroll
        for (int s = 0; s < 16; ++s)
          bK[s] = kp[(size_t)((s & 7) + ((s >> 3) << 4)) * S_LEN + 16 * j];
        sc[j] = wmma16(qf[t], bK, sc[j]);
      }
    }
    // ---- causal mask (only blocks crossing the diagonal) ----
    if (kb + 63 > q0) {
#pragma unroll
      for (int j = 0; j < 4; ++j) {
        const int key = kb + 16 * j + ll;
#pragma unroll
        for (int v = 0; v < 8; ++v) {
          const int row = q0 + v + 8 * lh;
          if (key > row) sc[j][v] = -1e30f;
        }
      }
    }
    // ---- online softmax: row max + rescale factor ----
    float mnew[8], alpha[8];
#pragma unroll
    for (int v = 0; v < 8; ++v) {
      float m = row_m[v];
#pragma unroll
      for (int j = 0; j < 4; ++j) m = fmaxf(m, sc[j][v]);
#pragma unroll
      for (int off = 1; off < 16; off <<= 1)
        m = fmaxf(m, __shfl_xor(m, off, 32));
      mnew[v] = m;
      alpha[v] = __expf(row_m[v] - m);
      row_m[v] = m;
    }
    // ---- P = exp(S - m): accumulate row sums, stage to LDS as f16 ----
    float ladd[8];
#pragma unroll
    for (int v = 0; v < 8; ++v) ladd[v] = 0.0f;
#pragma unroll
    for (int j = 0; j < 4; ++j)
#pragma unroll
      for (int v = 0; v < 8; ++v) {
        const float p = __expf(sc[j][v] - mnew[v]);
        ladd[v] += p;
        Pw[(v + 8 * lh) * 64 + 16 * j + ll] = (f16)p;
      }
#pragma unroll
    for (int v = 0; v < 8; ++v) {
      float s = ladd[v];
#pragma unroll
      for (int off = 1; off < 16; off <<= 1) s += __shfl_xor(s, off, 32);
      row_l[v] = row_l[v] * alpha[v] + s;
    }
    // DS ops are in-order within a wave; wait for the transpose staging.
    asm volatile("s_wait_dscnt 0x0" ::: "memory");
    // ---- reload P as A-fragments ----
    f16x16 pf[2];
#pragma unroll
    for (int t = 0; t < 2; ++t) {
      const f16* pp = Pw + ll * 64 + 32 * t + kb8;
      pf[t] = cat8(*(const f16x8*)pp, *(const f16x8*)(pp + 16));
    }
    // ---- rescale running output, then O += P * V ----
#pragma unroll
    for (int jd = 0; jd < 4; ++jd)
#pragma unroll
      for (int v = 0; v < 8; ++v) oacc[jd][v] *= alpha[v];
#pragma unroll
    for (int t = 0; t < 2; ++t) {
      const f16* vp = V + (size_t)(kb + 32 * t + kb8) * HD + ll;
#pragma unroll
      for (int jd = 0; jd < 4; ++jd) {
        f16x16 bV;
#pragma unroll
        for (int s = 0; s < 16; ++s)
          bV[s] = vp[(size_t)((s & 7) + ((s >> 3) << 4)) * HD + 16 * jd];
        oacc[jd] = wmma16(pf[t], bV, oacc[jd]);
      }
    }
  }
  // ---- normalize and store to O[B,S,H*HD] (f16, feeds output GEMM) ----
#pragma unroll
  for (int v = 0; v < 8; ++v) {
    const float inv_l = 1.0f / row_l[v];
    const int s = q0 + v + 8 * lh;
#pragma unroll
    for (int jd = 0; jd < 4; ++jd) {
      const int d = 16 * jd + ll;
      Ob[((size_t)(b * S_LEN + s)) * DM + h * HD + d] =
          (f16)(oacc[jd][v] * inv_l);
    }
  }
}

// ------------------------- Output projection -------------------------------
__global__ __launch_bounds__(256) void oproj_kernel(
    const f16* __restrict__ Ob, const f16* __restrict__ Wot,
    float* __restrict__ out) {
  __shared__ __attribute__((aligned(128))) f16 Bs[KPH * 64];   // 64 KB

  const int wave = threadIdx.x >> 5;
  const int lane = threadIdx.x & 31;
  const int ll = lane & 15;
  const int lh = lane >> 4;
  const int m0 = blockIdx.x * 256 + wave * 32;
  const int n0 = blockIdx.y * 64;

  v8f acc[2][4];
#pragma unroll
  for (int i = 0; i < 2; ++i)
#pragma unroll
    for (int j = 0; j < 4; ++j) acc[i][j] = (v8f)0.0f;

  gemm_panel(Ob, Wot, Bs, m0, n0, acc);

#pragma unroll
  for (int i = 0; i < 2; ++i)
#pragma unroll
    for (int j = 0; j < 4; ++j)
#pragma unroll
      for (int v = 0; v < 8; ++v) {
        const int m = m0 + 16 * i + v + 8 * lh;
        const int n = n0 + 16 * j + ll;
        out[(size_t)m * DM + n] = acc[i][j][v];
      }
}

// ------------------------------- Launcher ----------------------------------
extern "C" void kernel_launch(void* const* d_in, const int* in_sizes, int n_in,
                              void* d_out, int out_size, void* d_ws,
                              size_t ws_size, hipStream_t stream) {
  (void)in_sizes; (void)n_in; (void)out_size; (void)ws_size;
  const float* x  = (const float*)d_in[0];
  const float* Wq = (const float*)d_in[1];
  const float* Wk = (const float*)d_in[2];
  const float* Wv = (const float*)d_in[3];
  const float* Wo = (const float*)d_in[4];
  float* out = (float*)d_out;

  // Workspace carve-up (f16 elements). Total ~50.3 MB, resident in 192MB L2.
  f16* w = (f16*)d_ws;
  f16* Xb  = w; w += (size_t)MTOT * DM;     // 4096x1024
  f16* Wtq = w; w += (size_t)DM * DM;       // K-major weights
  f16* Wtk = w; w += (size_t)DM * DM;
  f16* Wtv = w; w += (size_t)DM * DM;
  f16* Wto = w; w += (size_t)DM * DM;
  f16* Qr  = w; w += (size_t)MTOT * DM;     // [B,H,S,HD]
  f16* Ktr = w; w += (size_t)MTOT * DM;     // [B,H,HD,S]
  f16* Vn  = w; w += (size_t)MTOT * DM;     // [B,H,S,HD]
  f16* Ob  = w; w += (size_t)MTOT * DM;     // [B,S,DM]

  const int NX = MTOT * DM;                 // 4194304
  cvt_f16_kernel<<<(NX + 255) / 256, 256, 0, stream>>>(x, Xb, NX);
  cvt_wt_kernel<<<(DM * DM) / 256, 256, 0, stream>>>(Wq, Wtq);
  cvt_wt_kernel<<<(DM * DM) / 256, 256, 0, stream>>>(Wk, Wtk);
  cvt_wt_kernel<<<(DM * DM) / 256, 256, 0, stream>>>(Wv, Wtv);
  cvt_wt_kernel<<<(DM * DM) / 256, 256, 0, stream>>>(Wo, Wto);

  // QKV projections + RoPE, head-aligned 64-wide N tiles
  qkv_kernel<<<dim3(MTOT / 256, DM / 64, 3), 256, 0, stream>>>(
      Xb, Wtq, Wtk, Wtv, Qr, Ktr, Vn);

  // Causal flash attention: (qtiles of 128) x (B*H heads)
  attn_kernel<<<dim3(S_LEN / 128, 2 * NH), 256, 0, stream>>>(Qr, Ktr, Vn, Ob);

  // Output projection -> f32
  oproj_kernel<<<dim3(MTOT / 256, DM / 64), 256, 0, stream>>>(Ob, Wto, out);
}
